// MyRNN_79379585564725
// MI455X (gfx1250) — matrix operand order repaired and checked by reading, not compiled, not run on hardware
//
#include <hip/hip_runtime.h>

// ---------------------------------------------------------------------------
// 2-layer SimpleRNN on gfx1250 (MI455X), wave32 + v_wmma_f32_16x16x32_f16.
//
// Formulation: compute transposed GEMMs  H'^T = W^T (64x64) x h^T (64x16)
// so the loop-invariant weights are WMMA A-operands held in registers, and
// the recurrent state h^T is the B-operand, round-tripped through LDS (f16,
// row-major by batch) for the C->B layout conversion each step.
// ---------------------------------------------------------------------------

typedef _Float16 v16h __attribute__((ext_vector_type(16)));
typedef _Float16 v8h  __attribute__((ext_vector_type(8)));
typedef float    v8f  __attribute__((ext_vector_type(8)));

#define VOCAB_ 10000
#define EMBED_ 100
#define SEQ_   80
#define UNITS_ 64
#define BATCH_ 16384
#define WAVES_ 4   // waves per block; each wave owns 16 batch rows

// ---------------- math helpers ----------------
__device__ __forceinline__ float fast_tanh(float x) {
#if __has_builtin(__builtin_amdgcn_tanhf)
  return __builtin_amdgcn_tanhf(x);              // gfx1250 v_tanh_f32
#elif __has_builtin(__builtin_amdgcn_exp2f)
  float e2 = __builtin_amdgcn_exp2f(x * 2.8853900817779268f);  // e^(2x)
  return (e2 - 1.0f) / (e2 + 1.0f);
#else
  return tanhf(x);
#endif
}

__device__ __forceinline__ float fast_sigmoid(float x) {
#if __has_builtin(__builtin_amdgcn_exp2f)
  float e = __builtin_amdgcn_exp2f(x * 1.4426950408889634f);   // e^x
  return e / (e + 1.0f);
#else
  return 1.0f / (1.0f + expf(-x));
#endif
}

// ---------------- WMMA fragment helpers ----------------
// A-fragment (16x32 f16) of W^T tile: lane L holds row m = 16*i + (L&15),
// element e maps to k = 32*c + 16*(L>>4) + e.
// W is stored row-major (k_in, m_out) with row stride 64, so W^T[m][k]=W[k*64+m].
__device__ __forceinline__ v16h load_wfrag(const float* __restrict__ W,
                                           int i, int c, int lane) {
  const int m  = 16 * i + (lane & 15);
  const int kb = 32 * c + 16 * (lane >> 4);
  v16h r;
#pragma unroll
  for (int e = 0; e < 16; ++e) {
    int k = kb + e;
    r[e] = (_Float16)W[k * UNITS_ + m];
  }
  return r;
}

// B-fragment (32x16 f16): lane L holds column n = L&15, contiguous
// k = 32*c + 16*(L>>4) .. +16  => 32 contiguous bytes in row-major (n,k) f16.
__device__ __forceinline__ v16h load_bfrag(const _Float16* p) {
  v16h r;
  ((int4*)&r)[0] = ((const int4*)p)[0];
  ((int4*)&r)[1] = ((const int4*)p)[1];
  return r;
}

// C-tile -> f16 store: lane L's v8f for m-tile i is batch row n = L&15,
// units 16*i + 8*(L>>4) + r, r = 0..7  => one contiguous 16-byte half8 store.
__device__ __forceinline__ void store_tile_f16(_Float16* dst, v8f a) {
  v8h p;
#pragma unroll
  for (int r = 0; r < 8; ++r) p[r] = (_Float16)a[r];
  *(int4*)dst = *(int4*)&p;
}

#define WMMA_F16(A, B, C) \
  __builtin_amdgcn_wmma_f32_16x16x32_f16(false, (A), false, (B), (short)0, (C), false, false)

// ---------------------------------------------------------------------------
// Kernel A: embProj[v][u] = sum_k emb[v][k] * Wx1[k][u] + b1[u]
// ---------------------------------------------------------------------------
__global__ void embproj_kernel(const float* __restrict__ emb,
                               const float* __restrict__ Wx1,
                               const float* __restrict__ b1,
                               float* __restrict__ embProj) {
  int idx = blockIdx.x * blockDim.x + threadIdx.x;
  if (idx >= VOCAB_ * UNITS_) return;
  int u = idx & (UNITS_ - 1);
  int v = idx >> 6;
  const float* er = emb + (long)v * EMBED_;
  float s = b1[u];
#pragma unroll 4
  for (int k = 0; k < EMBED_; ++k) s += er[k] * Wx1[k * UNITS_ + u];
  embProj[idx] = s;
}

// ---------------------------------------------------------------------------
// Kernel B: fused 2-layer recurrence + final dense/sigmoid.
// One wave per 16 batch rows; 80 serial steps; 24 WMMA per step per wave.
// ---------------------------------------------------------------------------
__global__ __launch_bounds__(WAVES_ * 32, 1)
void rnn_kernel(const int*   __restrict__ tokens,
                const float* __restrict__ embProj,  // VOCAB x 64, b1 folded in
                const float* __restrict__ Wh1,
                const float* __restrict__ Wx2,
                const float* __restrict__ Wh2,
                const float* __restrict__ b2,
                const float* __restrict__ Wd,
                const float* __restrict__ bd,
                float* __restrict__ out) {
  __shared__ _Float16 hbuf[WAVES_][2][16 * UNITS_];

  const int lane   = threadIdx.x & 31;
  const int wv     = threadIdx.x >> 5;
  const int half_  = lane >> 4;
  const int n      = lane & 15;                       // batch row within tile
  const int rowBase = (blockIdx.x * WAVES_ + wv) * 16;
  const int ub     = 8 * half_;                       // unit offset within 16-tile

  // ---- loop-invariant weight A-fragments (register resident) ----
  v16h A1[4][2], A2[4][2], A3[4][2];
#pragma unroll
  for (int i = 0; i < 4; ++i)
#pragma unroll
    for (int c = 0; c < 2; ++c) {
      A1[i][c] = load_wfrag(Wh1, i, c, lane);
      A2[i][c] = load_wfrag(Wx2, i, c, lane);
      A3[i][c] = load_wfrag(Wh2, i, c, lane);
    }

  // ---- b2 in C layout (initial accumulator for layer-2 WMMA chain) ----
  v8f b2f[4];
#pragma unroll
  for (int i = 0; i < 4; ++i)
#pragma unroll
    for (int r = 0; r < 8; ++r) b2f[i][r] = b2[16 * i + ub + r];

  _Float16* h1s = hbuf[wv][0];
  _Float16* h2s = hbuf[wv][1];

  v16h h1f[2], h2f[2];
#pragma unroll
  for (int c = 0; c < 2; ++c) { h1f[c] = (v16h)(_Float16)0.0f; h2f[c] = (v16h)(_Float16)0.0f; }

  const long tokrow = (long)(rowBase + n) * SEQ_;
  v8f acc2[4];

  for (int t = 0; t < SEQ_; ++t) {
    // ---- layer 1: acc1 = embProj[token]  (C-layout gather) + Wh1^T x h1^T ----
    int tok = tokens[tokrow + t];
    const float* ep = embProj + (long)tok * UNITS_;
    v8f acc1[4];
#pragma unroll
    for (int i = 0; i < 4; ++i) {
      const float4* p = (const float4*)(ep + 16 * i + ub);
      float4 lo = p[0], hi = p[1];
      acc1[i] = (v8f){lo.x, lo.y, lo.z, lo.w, hi.x, hi.y, hi.z, hi.w};
    }
#pragma unroll
    for (int i = 0; i < 4; ++i) {
      acc1[i] = WMMA_F16(A1[i][0], h1f[0], acc1[i]);
      acc1[i] = WMMA_F16(A1[i][1], h1f[1], acc1[i]);
    }
#pragma unroll
    for (int i = 0; i < 4; ++i) {
#pragma unroll
      for (int r = 0; r < 8; ++r) acc1[i][r] = fast_tanh(acc1[i][r]);
      store_tile_f16(h1s + n * UNITS_ + 16 * i + ub, acc1[i]);
    }
    asm volatile("" ::: "memory");  // keep ds_load after ds_store (same-wave LDS is in-order)
    h1f[0] = load_bfrag(h1s + n * UNITS_ + 0  + 16 * half_);
    h1f[1] = load_bfrag(h1s + n * UNITS_ + 32 + 16 * half_);

    // ---- layer 2: acc2 = b2 + Wx2^T x h1^T + Wh2^T x h2^T ----
#pragma unroll
    for (int i = 0; i < 4; ++i) {
      acc2[i] = b2f[i];
      acc2[i] = WMMA_F16(A2[i][0], h1f[0], acc2[i]);
      acc2[i] = WMMA_F16(A2[i][1], h1f[1], acc2[i]);
      acc2[i] = WMMA_F16(A3[i][0], h2f[0], acc2[i]);
      acc2[i] = WMMA_F16(A3[i][1], h2f[1], acc2[i]);
    }
#pragma unroll
    for (int i = 0; i < 4; ++i) {
#pragma unroll
      for (int r = 0; r < 8; ++r) acc2[i][r] = fast_tanh(acc2[i][r]);
      store_tile_f16(h2s + n * UNITS_ + 16 * i + ub, acc2[i]);
    }
    asm volatile("" ::: "memory");
    h2f[0] = load_bfrag(h2s + n * UNITS_ + 0  + 16 * half_);
    h2f[1] = load_bfrag(h2s + n * UNITS_ + 32 + 16 * half_);
  }

  // ---- final dense: out[b] = sigmoid(h2_last . Wd + bd) ----
  float partial = 0.0f;
#pragma unroll
  for (int i = 0; i < 4; ++i) {
    const float4* wp = (const float4*)(Wd + 16 * i + ub);
    float4 w0 = wp[0], w1 = wp[1];
    partial += acc2[i][0] * w0.x + acc2[i][1] * w0.y +
               acc2[i][2] * w0.z + acc2[i][3] * w0.w +
               acc2[i][4] * w1.x + acc2[i][5] * w1.y +
               acc2[i][6] * w1.z + acc2[i][7] * w1.w;
  }
  partial += __shfl_xor(partial, 16, 32);  // combine the two half-wave unit ranges
  if (lane < 16) out[rowBase + n] = fast_sigmoid(partial + bd[0]);
}

// ---------------------------------------------------------------------------
extern "C" void kernel_launch(void* const* d_in, const int* in_sizes, int n_in,
                              void* d_out, int out_size, void* d_ws, size_t ws_size,
                              hipStream_t stream) {
  const int*   tokens = (const int*)d_in[0];
  const float* emb    = (const float*)d_in[1];
  const float* Wx1    = (const float*)d_in[2];
  const float* Wh1    = (const float*)d_in[3];
  const float* b1     = (const float*)d_in[4];
  const float* Wx2    = (const float*)d_in[5];
  const float* Wh2    = (const float*)d_in[6];
  const float* b2     = (const float*)d_in[7];
  const float* Wd     = (const float*)d_in[8];
  const float* bd     = (const float*)d_in[9];
  float* out     = (float*)d_out;
  float* embProj = (float*)d_ws;  // VOCAB x 64 f32 = 2.56 MB scratch

  embproj_kernel<<<(VOCAB_ * UNITS_ + 255) / 256, 256, 0, stream>>>(emb, Wx1, b1, embProj);
  rnn_kernel<<<BATCH_ / (16 * WAVES_), WAVES_ * 32, 0, stream>>>(
      tokens, embProj, Wh1, Wx2, Wh2, b2, Wd, bd, out);
}
